// TransformerCAModule_51531017617995
// MI455X (gfx1250) — matrix operand order repaired and checked by reading, not compiled, not run on hardware
//
#include <hip/hip_runtime.h>
#include <hip/hip_bf16.h>
#include <math.h>

typedef _Float16 half_t;
typedef __attribute__((ext_vector_type(16))) _Float16 v16h;
typedef __attribute__((ext_vector_type(8)))  _Float16 v8h;
typedef __attribute__((ext_vector_type(8)))  float    v8f;
typedef __attribute__((ext_vector_type(4)))  unsigned int v4u;
typedef __attribute__((ext_vector_type(8)))  unsigned int v8u;

#define BB 4
#define HH 96
#define WW 96
#define NPIX (BB*HH*WW)      /* 36864 */
#define DM 128
#define TT 18
#define NWAVES 4608
#define NBLK 2304
#define BLKT 64
#define PITER (NPIX/NWAVES)  /* 8, exact */

/* shared TDM-staged weight buffer (one stage at a time, padded rows)      */
/* max stage: Wqkv 384 rows x (256B + 16B pad) = 104448 B                  */
#define WBUF_BYTES 104448
#define WSTR_128   136       /* halfs: 128-half row + 8 pad  (272 B pitch) */
#define WSTR_256   264       /* halfs: 256-half row + 8 pad  (528 B pitch) */

/* per-wave LDS slice layout (bytes, all 16B aligned) */
#define OFF_TOK   0        /* float [18][128]  residual stream   */
#define OFF_HBUF  9216     /* half  [32][128]  LN output (A src) */
#define OFF_QBUF  17408    /* half  [32][128]  Q                 */
#define OFF_KBUF  25600    /* half  [32][128]  K                 */
#define OFF_VTB   33792    /* half  [128][32]  V transposed      */
#define OFF_OBUF  41984    /* half  [32][128]  attn output       */
#define OFF_H1    50176    /* half  [32][256]  FFN hidden        */
#define OFF_SBUF  66560    /* float [32][32]   scores            */
#define OFF_PBUF  70656    /* half  [32][32]   probs (A src)     */
#define WAVE_LDS  72704

#define TOTAL_LDS (WBUF_BYTES + 2*WAVE_LDS)   /* 249856 B < 320 KB */

struct LayerArgs {
  const float *ln1_g, *ln1_b, *bqkv, *bo, *ln2_g, *ln2_b, *b1, *b2;
  const half_t *wqkv, *wo, *w1, *w2;   /* f16 copies in workspace */
};
struct KArgs {
  const half_t* xe;      /* [NPIX][128] f16 embedded pixels */
  const float*  pos;     /* [18][128] */
  LayerArgs L[2];
  const float *lnf_g, *lnf_b, *hw0, *hw1, *hb0, *hb1;
  float* out;            /* [2][NPIX] */
};

/* ---- WMMA fragment loaders (CDNA5 16-bit layouts, wave32) ---- */
/* A 16x32: lane<16 holds row m0+lane, K {k0+0..7, k0+16..23}; lane>=16: K {k0+8..15, k0+24..31} */
__device__ __forceinline__ v16h load_a(const half_t* A, int stride, int m0, int k0, int lane){
  int hf = lane >> 4, r = lane & 15;
  const half_t* p0 = A + (m0 + r)*stride + k0 + hf*8;
  v8h lo = *(const v8h*)p0;
  v8h hi = *(const v8h*)(p0 + 16);
  v16h a;
#pragma unroll
  for (int i=0;i<8;++i){ a[i]=lo[i]; a[8+i]=hi[i]; }
  return a;
}
/* B 32x16 from row-major W[n][k]: lane<16 holds col n0+lane, K {k0+0..7, k0+16..23}; lane>=16: K {k0+8..15, k0+24..31} */
__device__ __forceinline__ v16h load_b(const half_t* Bm, int stride, int n0, int k0, int lane){
  int hf = lane >> 4, c = lane & 15;
  const half_t* p0 = Bm + (n0 + c)*stride + k0 + hf*8;
  v8h lo = *(const v8h*)p0;
  v8h hi = *(const v8h*)(p0 + 16);
  v16h b;
#pragma unroll
  for (int i=0;i<8;++i){ b[i]=lo[i]; b[8+i]=hi[i]; }
  return b;
}

template<int KS>
__device__ __forceinline__ v8f gemm_tile(const half_t* A, int astr, int m0,
                                         const half_t* Bm, int bstr, int n0,
                                         int k0, int lane){
  v8f acc = {};
#pragma unroll
  for (int kk=0;kk<KS;++kk){
    v16h a = load_a(A, astr, m0, k0 + kk*32, lane);
    v16h b = load_b(Bm, bstr, n0, k0 + kk*32, lane);
    acc = __builtin_amdgcn_wmma_f32_16x16x32_f16(false, a, false, b, (short)0, acc, false, false);
  }
  return acc;
}

__device__ __forceinline__ float wave_sum(float v){
#pragma unroll
  for (int o=16;o>=1;o>>=1) v += __shfl_xor(v, o, 32);
  return v;
}

/* LayerNorm rows 0..17 of tokf (f32) -> dst (f16), wave cooperative */
__device__ __forceinline__ void ln_rows(const float* tokf, half_t* dst,
                                        const float* g, const float* b, int lane){
  for (int t=0;t<TT;++t){
    float s=0.f, ss=0.f;
#pragma unroll
    for (int q=0;q<4;++q){ float v = tokf[t*DM + lane*4 + q]; s += v; ss += v*v; }
    s = wave_sum(s); ss = wave_sum(ss);
    float m = s*(1.f/DM);
    float rstd = rsqrtf(ss*(1.f/DM) - m*m + 1e-5f);
#pragma unroll
    for (int q=0;q<4;++q){
      int d = lane*4+q;
      dst[t*DM+d] = (half_t)((tokf[t*DM+d]-m)*rstd*g[d] + b[d]);
    }
  }
}

/* ---- Tensor Data Mover: async copy one weight matrix (global f16 -> LDS) ----
 * 2D descriptor, data_size = 4B dwords, row = elemPerRow dwords,
 * pad_enable adds 4 dwords (16B) after each row so LDS pitch rotates banks.
 * LDS dest = offset 0 of dynamic LDS (the shared weight buffer).
 * intervalCode: 5 -> pad every 64 dwords, 6 -> every 128 dwords.
 */
__device__ __forceinline__ void tdm_load_w(const half_t* gsrc, int elemPerRow,
                                           int nrows, int intervalCode){
  unsigned long long ga = (unsigned long long)(uintptr_t)gsrc;
  v4u g0; v8u g1; v4u g2 = {0u,0u,0u,0u}; v4u g3 = {0u,0u,0u,0u};
  g0[0] = 1u;                                            /* count=1, user desc */
  g0[1] = 0u;                                            /* lds_addr = wbuf    */
  g0[2] = (unsigned int)ga;                              /* global_addr[31:0]  */
  g0[3] = ((unsigned int)(ga >> 32) & 0x01FFFFFFu) | 0x80000000u; /* +type=2  */
  g1[0] = (2u << 16) | (1u << 20) | ((unsigned int)intervalCode << 22) | (3u << 25);
  g1[1] = ((unsigned int)elemPerRow & 0xFFFFu) << 16;    /* tensor_dim0 lo     */
  g1[2] = (((unsigned int)elemPerRow >> 16) & 0xFFFFu)   /* tensor_dim0 hi     */
        | (((unsigned int)nrows & 0xFFFFu) << 16);       /* tensor_dim1 lo     */
  g1[3] = ((unsigned int)elemPerRow & 0xFFFFu) << 16;    /* tile_dim0          */
  g1[4] = (unsigned int)nrows & 0xFFFFu;                 /* tile_dim1          */
  g1[5] = (unsigned int)elemPerRow;                      /* dim0_stride lo     */
  g1[6] = 0u;
  g1[7] = 0u;
  asm volatile("tensor_load_to_lds %0, %1, %2, %3"
               :: "s"(g0), "s"(g1), "s"(g2), "s"(g3) : "memory");
}

/* ---------------- pre-pass kernels ---------------- */
__global__ void cvt_f32_f16(const float* __restrict__ s, half_t* __restrict__ d, int n){
  int i = blockIdx.x*blockDim.x + threadIdx.x;
  if (i < n) d[i] = (half_t)s[i];
}

/* 1x1 conv embedding: xe[pix][d] = sum_c x[b,c,h,w]*emb_w[d][c] + emb_b[d] */
__global__ void embed_kernel(const float* __restrict__ x, const float* __restrict__ ew,
                             const float* __restrict__ eb, half_t* __restrict__ xe){
  __shared__ float xc[16];
  int p = blockIdx.x;
  int b = p/(HH*WW), rem = p%(HH*WW), h = rem/WW, w = rem%WW;
  int d = threadIdx.x;
  if (d < 16) xc[d] = x[((b*16 + d)*HH + h)*WW + w];
  __syncthreads();
  float acc = eb[d];
#pragma unroll
  for (int c=0;c<16;++c) acc += xc[c]*ew[d*16 + c];
  xe[(size_t)p*DM + d] = (half_t)acc;
}

/* ---------------- fused 2-layer transformer, 1 wave = 1 pixel ---------------- */
__global__ void transformer_kernel(KArgs P){
  extern __shared__ char smem[];
  const int tid  = threadIdx.x;
  const int wave = tid >> 5;
  const int lane = tid & 31;
  half_t* wbuf = (half_t*)smem;                 /* shared staged weights */
  char* base = smem + WBUF_BYTES + wave*WAVE_LDS;
  float*  tokf = (float*)(base + OFF_TOK);
  half_t* hbuf = (half_t*)(base + OFF_HBUF);
  half_t* qbuf = (half_t*)(base + OFF_QBUF);
  half_t* kbuf = (half_t*)(base + OFF_KBUF);
  half_t* vtb  = (half_t*)(base + OFF_VTB);
  half_t* obuf = (half_t*)(base + OFF_OBUF);
  half_t* h1   = (half_t*)(base + OFF_H1);
  float*  sbuf = (float*)(base + OFF_SBUF);
  half_t* pbuf = (half_t*)(base + OFF_PBUF);
  const int gw = blockIdx.x*2 + wave;

  for (int it=0; it<PITER; ++it){
    const int pixel = it*NWAVES + gw;
    const int b  = pixel/(HH*WW);
    const int rm = pixel%(HH*WW);
    const int ph = rm/WW, pw = rm%WW;
    if (it+1 < PITER)
      __builtin_prefetch((const void*)(P.xe + (size_t)(pixel + NWAVES)*DM), 0, 0);

    /* gather 18 neighborhood tokens (zero-pad OOB) + positional embedding */
    for (int t=0;t<TT;++t){
      int dil = (t<9)?1:2;
      int r9  = (t<9)?t:(t-9);
      int di = r9/3 - 1, dj = r9%3 - 1;
      int sh = ph + di*dil, sw = pw + dj*dil;
      bool in = (sh>=0) && (sh<HH) && (sw>=0) && (sw<WW);   /* wave-uniform */
      long so = ((long)(b*HH + sh)*WW + sw)*DM;
#pragma unroll
      for (int q=0;q<4;++q){
        int d = lane*4+q;
        float v = in ? (float)P.xe[so + d] : 0.f;
        tokf[t*DM+d] = v + P.pos[t*DM+d];
      }
    }
    /* zero V-transpose (K-dim padding for attn@V must be 0) */
    for (int idx=lane; idx<128*32; idx+=32) vtb[idx] = (half_t)0;
    __syncthreads();

    for (int li=0; li<2; ++li){
      const LayerArgs& L = P.L[li];

      /* stage Wqkv via TDM while doing LN1 */
      if (wave == 0) tdm_load_w(L.wqkv, 64, 384, 5);
      ln_rows(tokf, hbuf, L.ln1_g, L.ln1_b, lane);
      if (wave == 0) __builtin_amdgcn_s_wait_tensorcnt((short)0);
      __syncthreads();

      /* ---- QKV: [18,128] x [128,384]^T (B from LDS wbuf) ---- */
      for (int mt=0;mt<2;++mt)
        for (int nt=0;nt<24;++nt){
          v8f acc = gemm_tile<4>(hbuf, DM, mt*16, wbuf, WSTR_128, nt*16, 0, lane);
          int col = nt*16 + (lane&15);
          float bias = L.bqkv[col];
#pragma unroll
          for (int j=0;j<8;++j){
            int row = mt*16 + j + ((lane>>4)<<3);
            if (row < TT){
              float v = acc[j] + bias;
              if (col < 128)      qbuf[row*DM + col]        = (half_t)v;
              else if (col < 256) kbuf[row*DM + (col-128)]  = (half_t)v;
              else                vtb[(col-256)*32 + row]   = (half_t)v;
            }
          }
        }
      __syncthreads();

      /* stage Wo via TDM while doing attention (attention never touches wbuf) */
      if (wave == 0) tdm_load_w(L.wo, 64, 128, 5);

      /* ---- attention, per head ---- */
      for (int hh=0; hh<4; ++hh){
        /* scores = Q K^T / sqrt(32) */
        for (int mt=0;mt<2;++mt)
          for (int nt=0;nt<2;++nt){
            v8f acc = gemm_tile<1>(qbuf, DM, mt*16, kbuf, DM, nt*16, hh*32, lane);
#pragma unroll
            for (int j=0;j<8;++j){
              int row = mt*16 + j + ((lane>>4)<<3);
              sbuf[row*32 + nt*16 + (lane&15)] = acc[j]*0.17677669529663687f;
            }
          }
        __syncthreads();
        /* softmax rows (one lane per token row) */
        if (lane < TT){
          float mx = -1e30f;
          for (int c=0;c<TT;++c) mx = fmaxf(mx, sbuf[lane*32+c]);
          float ex[TT]; float s = 0.f;
          for (int c=0;c<TT;++c){ ex[c] = expf(sbuf[lane*32+c]-mx); s += ex[c]; }
          float inv = 1.f/s;
          for (int c=0;c<TT;++c) pbuf[lane*32+c] = (half_t)(ex[c]*inv);
          for (int c=TT;c<32;++c) pbuf[lane*32+c] = (half_t)0;  /* K-pad */
        }
        __syncthreads();
        /* attn @ V (K padded to 32 with zeros) */
        for (int mt=0;mt<2;++mt)
          for (int nt=0;nt<2;++nt){
            v8f acc = gemm_tile<1>(pbuf, 32, mt*16, vtb, 32, nt*16, 0, lane);
#pragma unroll
            for (int j=0;j<8;++j){
              int row = mt*16 + j + ((lane>>4)<<3);
              if (row < TT) obuf[row*DM + hh*32 + nt*16 + (lane&15)] = (half_t)acc[j];
            }
          }
        __syncthreads();
      }

      if (wave == 0) __builtin_amdgcn_s_wait_tensorcnt((short)0);
      __syncthreads();

      /* ---- out-proj + residual (B from wbuf) ---- */
      for (int mt=0;mt<2;++mt)
        for (int nt=0;nt<8;++nt){
          v8f acc = gemm_tile<4>(obuf, DM, mt*16, wbuf, WSTR_128, nt*16, 0, lane);
          int col = nt*16 + (lane&15);
          float bias = L.bo[col];
#pragma unroll
          for (int j=0;j<8;++j){
            int row = mt*16 + j + ((lane>>4)<<3);
            if (row < TT) tokf[row*DM + col] += acc[j] + bias;
          }
        }
      __syncthreads();

      /* stage W1 via TDM while doing LN2 */
      if (wave == 0) tdm_load_w(L.w1, 64, 256, 5);
      ln_rows(tokf, hbuf, L.ln2_g, L.ln2_b, lane);
      if (wave == 0) __builtin_amdgcn_s_wait_tensorcnt((short)0);
      __syncthreads();

      /* ---- FFN1 + exact GELU (B from wbuf) ---- */
      for (int mt=0;mt<2;++mt)
        for (int nt=0;nt<16;++nt){
          v8f acc = gemm_tile<4>(hbuf, DM, mt*16, wbuf, WSTR_128, nt*16, 0, lane);
          int col = nt*16 + (lane&15);
          float bias = L.b1[col];
#pragma unroll
          for (int j=0;j<8;++j){
            int row = mt*16 + j + ((lane>>4)<<3);
            if (row < TT){
              float xg = acc[j] + bias;
              float gl = 0.5f*xg*(1.f + erff(xg*0.70710678118654752f));
              h1[row*256 + col] = (half_t)gl;
            }
          }
        }
      __syncthreads();

      /* stage W2 via TDM */
      if (wave == 0){
        tdm_load_w(L.w2, 128, 128, 6);
        __builtin_amdgcn_s_wait_tensorcnt((short)0);
      }
      __syncthreads();

      /* ---- FFN2 + residual (B from wbuf, 256-K) ---- */
      for (int mt=0;mt<2;++mt)
        for (int nt=0;nt<8;++nt){
          v8f acc = gemm_tile<8>(h1, 256, mt*16, wbuf, WSTR_256, nt*16, 0, lane);
          int col = nt*16 + (lane&15);
          float bias = L.b2[col];
#pragma unroll
          for (int j=0;j<8;++j){
            int row = mt*16 + j + ((lane>>4)<<3);
            if (row < TT) tokf[row*DM + col] += acc[j] + bias;
          }
        }
      __syncthreads();

      /* ---- final LN on center token (t=4) + linear head ---- */
      {
        float s=0.f, ss=0.f;
#pragma unroll
        for (int q=0;q<4;++q){ float v = tokf[4*DM + lane*4 + q]; s += v; ss += v*v; }
        s = wave_sum(s); ss = wave_sum(ss);
        float m = s*(1.f/DM);
        float rstd = rsqrtf(ss*(1.f/DM) - m*m + 1e-5f);
        const float* hw = (li==0) ? P.hw0 : P.hw1;
        float dp = 0.f;
#pragma unroll
        for (int q=0;q<4;++q){
          int d = lane*4+q;
          float c = (tokf[4*DM+d]-m)*rstd*P.lnf_g[d] + P.lnf_b[d];
          dp += c*hw[d];
        }
        dp = wave_sum(dp);
        if (lane == 0){
          float hb = (li==0) ? P.hb0[0] : P.hb1[0];
          P.out[(size_t)li*NPIX + pixel] = dp + hb;
        }
      }
      __syncthreads();
    }
  }
}

extern "C" void kernel_launch(void* const* d_in, const int* in_sizes, int n_in,
                              void* d_out, int out_size, void* d_ws, size_t ws_size,
                              hipStream_t stream){
  (void)in_sizes; (void)n_in; (void)out_size; (void)ws_size;
  /* input order (setup_inputs dict order, recursive):
     0:x 1:emb_w 2:emb_b 3:pos
     per layer (base 4 + 12*li): ln1_g ln1_b Wqkv bqkv Wo bo ln2_g ln2_b W1 b1 W2 b2
     28:lnf_g 29:lnf_b 30:head_w0 31:head_w1 32:head_b0 33:head_b1 */
  const float* x     = (const float*)d_in[0];
  const float* emb_w = (const float*)d_in[1];
  const float* emb_b = (const float*)d_in[2];

  half_t* xe  = (half_t*)d_ws;                 /* 36864*128 f16 = 9.4MB */
  half_t* wts = xe + (size_t)NPIX*DM;          /* 2*131072 f16  = 0.5MB */

  KArgs P;
  P.xe  = xe;
  P.pos = (const float*)d_in[3];
  P.out = (float*)d_out;
  for (int li=0; li<2; ++li){
    int ib = 4 + 12*li;
    LayerArgs& L = P.L[li];
    L.ln1_g = (const float*)d_in[ib+0];
    L.ln1_b = (const float*)d_in[ib+1];
    const float* wqkv_s = (const float*)d_in[ib+2];
    L.bqkv  = (const float*)d_in[ib+3];
    const float* wo_s   = (const float*)d_in[ib+4];
    L.bo    = (const float*)d_in[ib+5];
    L.ln2_g = (const float*)d_in[ib+6];
    L.ln2_b = (const float*)d_in[ib+7];
    const float* w1_s   = (const float*)d_in[ib+8];
    L.b1    = (const float*)d_in[ib+9];
    const float* w2_s   = (const float*)d_in[ib+10];
    L.b2    = (const float*)d_in[ib+11];
    half_t* lb = wts + (size_t)li*131072;
    L.wqkv = lb; L.wo = lb + 49152; L.w1 = lb + 65536; L.w2 = lb + 98304;
    cvt_f32_f16<<<(49152+255)/256, 256, 0, stream>>>(wqkv_s, lb,         49152);
    cvt_f32_f16<<<(16384+255)/256, 256, 0, stream>>>(wo_s,   lb + 49152, 16384);
    cvt_f32_f16<<<(32768+255)/256, 256, 0, stream>>>(w1_s,   lb + 65536, 32768);
    cvt_f32_f16<<<(32768+255)/256, 256, 0, stream>>>(w2_s,   lb + 98304, 32768);
  }
  P.lnf_g = (const float*)d_in[28];
  P.lnf_b = (const float*)d_in[29];
  P.hw0   = (const float*)d_in[30];
  P.hw1   = (const float*)d_in[31];
  P.hb0   = (const float*)d_in[32];
  P.hb1   = (const float*)d_in[33];

  embed_kernel<<<NPIX, 128, 0, stream>>>(x, emb_w, emb_b, xe);
  transformer_kernel<<<NBLK, BLKT, TOTAL_LDS, stream>>>(P);
}
